// EfficientMultiHotEmbedding_81054622810391
// MI455X (gfx1250) — matrix operand order repaired and checked by reading, not compiled, not run on hardware
//
#include <hip/hip_runtime.h>

typedef __attribute__((ext_vector_type(16))) _Float16 v16h;
typedef __attribute__((ext_vector_type(8)))  float    v8f;
typedef __attribute__((ext_vector_type(4)))  int      v4i;

#define NFEAT   10000
#define EDIM    64
#define NBATCH  16384
#define KTILES  313              // ceil(NFEAT / 32)
#define KSPLIT  8
#define TPS     40               // ceil(KTILES / KSPLIT)

// Unroll-by-2 requires every wave's full-tile count to be even:
// waves 0..KSPLIT-2 run TPS full tiles; last wave runs (KTILES-1) - (KSPLIT-1)*TPS.
static_assert(TPS % 2 == 0, "per-wave tile count must be even");
static_assert(((KTILES - 1) - (KSPLIT - 1) * TPS) % 2 == 0, "last wave tile count must be even");
static_assert((KTILES - 1) - (KSPLIT - 1) * TPS >= 2, "last wave needs >= 2 full tiles");

// ---------------------------------------------------------------------------
// Kernel 1: repack embed[NFEAT,64] f32 -> f16 WMMA B-fragments in workspace.
// bpack[(((t*4 + ntile)*32 + lane)*16) + i] where
//   K = 32*t + 8*(lane>>4) + (i&7) + 16*(i>>3)   (B 32x16: lanes 0-15 K=0-7,16-23, lanes 16-31 K=8-15,24-31)
//   N = ntile*16 + (lane&15)
// K >= NFEAT zero-padded so the main-kernel tail needs no B guard.
// ---------------------------------------------------------------------------
__global__ void pack_b_kernel(const float* __restrict__ embed,
                              _Float16* __restrict__ bpack) {
  int idx = blockIdx.x * blockDim.x + threadIdx.x;
  if (idx >= KTILES * 4 * 32 * 16) return;
  int i     = idx & 15;
  int lane  = (idx >> 4) & 31;
  int ntile = (idx >> 9) & 3;
  int t     = idx >> 11;
  int k = 32 * t + 8 * (lane >> 4) + (i & 7) + 16 * (i >> 3);
  int n = ntile * 16 + (lane & 15);
  float v = (k < NFEAT) ? embed[(size_t)k * EDIM + n] : 0.0f;
  bpack[idx] = (_Float16)v;
}

// ---------------------------------------------------------------------------
// Kernel 2: one 16-row M-tile per block; 8 waves split K 8 ways;
// unroll-by-2 ping-pong pipeline; deterministic LDS reduce.
// ---------------------------------------------------------------------------
__global__ __launch_bounds__(256)
void mhe_wmma_kernel(const int* __restrict__ mh,
                     const _Float16* __restrict__ bpack,
                     float* __restrict__ out) {
  __shared__ float red[KSPLIT][32][32];   // [wave][slot j][lane] -> conflict-free

  const int lane  = threadIdx.x & 31;
  const int wave  = threadIdx.x >> 5;     // k-split id 0..7
  const int half  = lane >> 4;
  const int row   = lane & 15;
  const int mtile = blockIdx.x;

  const int* __restrict__ arow = mh + (size_t)(mtile * 16 + row) * NFEAT;
  const _Float16* __restrict__ bbase = bpack + lane * 16;

  v8f acc0 = {}, acc1 = {}, acc2 = {}, acc3 = {};

  const int tb    = wave * TPS;
  const int te    = (tb + TPS < KTILES) ? (tb + TPS) : KTILES;
  const int nfull = ((te < KTILES - 1) ? te : (KTILES - 1)) - tb;   // 40 or 32, even

  // streaming pointers (A: +128B/tile per lane; B: +4KB/tile per lane group)
  const int*      ap = arow + 32 * tb + 8 * half;
  const _Float16* bp = bbase + (size_t)tb * 2048;

  // ping-pong fragment buffers
  v4i  xa0, xa1, xa2, xa3;  v16h xb0, xb1, xb2, xb3;   // X
  v4i  ya0, ya1, ya2, ya3;  v16h yb0, yb1, yb2, yb3;   // Y

#define LOAD_FRAGS(A0,A1,A2,A3, B0,B1,B2,B3, APTR, BPTR)                    \
  do {                                                                      \
    A0 = *(const v4i*)((APTR));        A1 = *(const v4i*)((APTR) + 4);      \
    A2 = *(const v4i*)((APTR) + 16);   A3 = *(const v4i*)((APTR) + 20);     \
    B0 = *(const v16h*)((BPTR));        B1 = *(const v16h*)((BPTR) + 512);  \
    B2 = *(const v16h*)((BPTR) + 1024); B3 = *(const v16h*)((BPTR) + 1536); \
  } while (0)

#define CONSUME(A0,A1,A2,A3, B0,B1,B2,B3)                                            \
  do {                                                                               \
    v16h a_;                                                                         \
    _Pragma("unroll")                                                                \
    for (int j = 0; j < 4; ++j) {                                                    \
      a_[j]      = (_Float16)(short)A0[j];   /* 0/1 exact in f16 */                  \
      a_[j + 4]  = (_Float16)(short)A1[j];                                           \
      a_[j + 8]  = (_Float16)(short)A2[j];                                           \
      a_[j + 12] = (_Float16)(short)A3[j];                                           \
    }                                                                                \
    acc0 = __builtin_amdgcn_wmma_f32_16x16x32_f16(false, a_, false, B0, (short)0, acc0, false, false); \
    acc1 = __builtin_amdgcn_wmma_f32_16x16x32_f16(false, a_, false, B1, (short)0, acc1, false, false); \
    acc2 = __builtin_amdgcn_wmma_f32_16x16x32_f16(false, a_, false, B2, (short)0, acc2, false, false); \
    acc3 = __builtin_amdgcn_wmma_f32_16x16x32_f16(false, a_, false, B3, (short)0, acc3, false, false); \
  } while (0)

  // ---- prologue: X <- tile tb ----
  LOAD_FRAGS(xa0, xa1, xa2, xa3, xb0, xb1, xb2, xb3, ap, bp);

  // ---- steady state: 2 tiles per iteration, no buffer rotation ----
  int s;
  for (s = 0; s + 2 < nfull; s += 2) {
    LOAD_FRAGS(ya0, ya1, ya2, ya3, yb0, yb1, yb2, yb3, ap + 32, bp + 2048);
    if (s + 8 < nfull) __builtin_prefetch(ap + 256, 0, 0);       // 1KB ahead, in-range
    CONSUME(xa0, xa1, xa2, xa3, xb0, xb1, xb2, xb3);             // tile tb+s

    ap += 64; bp += 4096;
    LOAD_FRAGS(xa0, xa1, xa2, xa3, xb0, xb1, xb2, xb3, ap, bp);  // tile tb+s+2
    if (s + 9 < nfull) __builtin_prefetch(ap + 224, 0, 0);
    CONSUME(ya0, ya1, ya2, ya3, yb0, yb1, yb2, yb3);             // tile tb+s+1
  }

  // ---- epilogue: last two full tiles (s == nfull-2) ----
  LOAD_FRAGS(ya0, ya1, ya2, ya3, yb0, yb1, yb2, yb3, ap + 32, bp + 2048);
  CONSUME(xa0, xa1, xa2, xa3, xb0, xb1, xb2, xb3);
  CONSUME(ya0, ya1, ya2, ya3, yb0, yb1, yb2, yb3);

  if (te == KTILES) {           // tail tile: K 9984..10015, guard A past NFEAT
    const int c0 = 32 * (KTILES - 1) + 8 * half;
    v16h a;
#pragma unroll
    for (int j = 0; j < 8; ++j) {
      int k0 = c0 + j;
      int k1 = c0 + 16 + j;
      a[j]     = (k0 < NFEAT) ? (_Float16)(short)arow[k0] : (_Float16)0.0f;
      a[j + 8] = (k1 < NFEAT) ? (_Float16)(short)arow[k1] : (_Float16)0.0f;
    }
    const _Float16* bt = bbase + (size_t)(KTILES - 1) * 2048;
    v16h t0 = *(const v16h*)(bt);
    v16h t1 = *(const v16h*)(bt + 512);
    v16h t2 = *(const v16h*)(bt + 1024);
    v16h t3 = *(const v16h*)(bt + 1536);
    acc0 = __builtin_amdgcn_wmma_f32_16x16x32_f16(false, a, false, t0, (short)0, acc0, false, false);
    acc1 = __builtin_amdgcn_wmma_f32_16x16x32_f16(false, a, false, t1, (short)0, acc1, false, false);
    acc2 = __builtin_amdgcn_wmma_f32_16x16x32_f16(false, a, false, t2, (short)0, acc2, false, false);
    acc3 = __builtin_amdgcn_wmma_f32_16x16x32_f16(false, a, false, t3, (short)0, acc3, false, false);
  }

  // Deterministic K reduction across the 8 waves via LDS.
  if (wave != 0) {
#pragma unroll
    for (int v = 0; v < 8; ++v) {
      red[wave][v][lane]      = acc0[v];
      red[wave][8 + v][lane]  = acc1[v];
      red[wave][16 + v][lane] = acc2[v];
      red[wave][24 + v][lane] = acc3[v];
    }
  }
  __syncthreads();

  if (wave == 0) {
#pragma unroll
    for (int w = 1; w < KSPLIT; ++w) {
#pragma unroll
      for (int v = 0; v < 8; ++v) {
        acc0[v] += red[w][v][lane];
        acc1[v] += red[w][8 + v][lane];
        acc2[v] += red[w][16 + v][lane];
        acc3[v] += red[w][24 + v][lane];
      }
    }
    // D layout: VGPR v -> M = 8*half + v, N = lane&15 (+16 per ntile)
#pragma unroll
    for (int v = 0; v < 8; ++v) {
      size_t m = (size_t)mtile * 16 + 8 * half + v;
      float* o = out + m * EDIM + row;
      o[0]  = acc0[v];
      o[16] = acc1[v];
      o[32] = acc2[v];
      o[48] = acc3[v];
    }
  }

#undef LOAD_FRAGS
#undef CONSUME
}

extern "C" void kernel_launch(void* const* d_in, const int* in_sizes, int n_in,
                              void* d_out, int out_size, void* d_ws, size_t ws_size,
                              hipStream_t stream) {
  const int*   mh    = (const int*)d_in[0];     // multi_hot [16384,10000] int32
  const float* embed = (const float*)d_in[1];   // embed     [10000,64]   f32
  float*       out   = (float*)d_out;           // out       [16384,64]   f32
  _Float16*    bpack = (_Float16*)d_ws;         // 313*4*32*16 f16 = 1.22 MB

  const int packN = KTILES * 4 * 32 * 16;
  pack_b_kernel<<<(packN + 255) / 256, 256, 0, stream>>>(embed, bpack);
  mhe_wmma_kernel<<<NBATCH / 16, 256, 0, stream>>>(mh, bpack, out);
}
